// GraphAttentionLayer_53532472377399
// MI455X (gfx1250) — matrix operand order repaired and checked by reading, not compiled, not run on hardware
//
#include <hip/hip_runtime.h>
#include <hip/hip_bf16.h>

// Problem constants (from the reference)
#define NN   50000
#define EE   800000
#define H    8
#define HD   32
#define DIM  256   // IN_DIM == OUT_DIM

typedef __attribute__((ext_vector_type(16))) __bf16 v16bf;
typedef __attribute__((ext_vector_type(8)))  __bf16 v8bf;
typedef __attribute__((ext_vector_type(8)))  float  v8f;

__device__ __forceinline__ unsigned short f2bf(float f) {
    unsigned int u = __float_as_uint(f);
    u += 0x7FFFu + ((u >> 16) & 1u);   // round-to-nearest-even
    return (unsigned short)(u >> 16);
}

// ---------------------------------------------------------------------------
// Prep kernels: zero score/denominator buffers, bf16-convert x, transpose+
// convert W_src/W_dst, and initialize out = x (residual folded into scatter).
// ---------------------------------------------------------------------------
__global__ __launch_bounds__(256)
void k_zero(float* __restrict__ p, int n) {
    int i = blockIdx.x * 256 + threadIdx.x;
    if (i < n) p[i] = 0.0f;
}

__global__ __launch_bounds__(256)
void k_cvt_x(const float* __restrict__ x, unsigned short* __restrict__ xb, int n4) {
    int i = blockIdx.x * 256 + threadIdx.x;
    if (i >= n4) return;
    float4 v = *(const float4*)(x + (size_t)i * 4);
    ushort4 o;
    o.x = f2bf(v.x); o.y = f2bf(v.y); o.z = f2bf(v.z); o.w = f2bf(v.w);
    *(ushort4*)(xb + (size_t)i * 4) = o;
}

__global__ __launch_bounds__(256)
void k_cvt_w(const float* __restrict__ Wsrc, const float* __restrict__ Wdst,
             unsigned short* __restrict__ wt) {
    int t = blockIdx.x * 256 + threadIdx.x;       // t < 2*256*256
    if (t >= 2 * DIM * DIM) return;
    int z   = t >> 16;
    int idx = t & 0xFFFF;
    int k   = idx & (DIM - 1);
    int n   = idx >> 8;
    const float* W = z ? Wdst : Wsrc;
    wt[(size_t)z * DIM * DIM + (size_t)n * DIM + k] = f2bf(W[(size_t)k * DIM + n]);
}

__global__ __launch_bounds__(256)
void k_init_out(const float* __restrict__ x, float* __restrict__ out, int n4) {
    int i = blockIdx.x * 256 + threadIdx.x;
    if (i >= n4) return;
    *(float4*)(out + (size_t)i * 4) = *(const float4*)(x + (size_t)i * 4);
}

// ---------------------------------------------------------------------------
// Fused projection GEMM + per-node attention scores.
// Block = 512 threads = 16 wave32 waves. Wave w computes the 16x16 tile at
// rows [16*blockIdx.x, +16), cols [16*w, +16). grid.z selects src/dst matrix.
// K is consumed in 8 steps of 32 via v_wmma_f32_16x16x32_bf16.
// ---------------------------------------------------------------------------
__global__ __launch_bounds__(512, 2)
void k_gemm_proj(const unsigned short* __restrict__ xb,
                 const unsigned short* __restrict__ wt,
                 const float* __restrict__ b_src, const float* __restrict__ b_dst,
                 const float* __restrict__ attn,
                 float* __restrict__ proj_src, float* __restrict__ proj_dst,
                 float* __restrict__ s_src, float* __restrict__ s_dst) {
    const int z    = blockIdx.z;
    const int m0   = blockIdx.x << 4;      // N = 50000 = 3125 * 16 exactly
    const int wave = threadIdx.x >> 5;
    const int lane = threadIdx.x & 31;
    const int half = lane >> 4;
    const int nl   = lane & 15;
    const int n0   = wave << 4;

    // C init = bias broadcast (C layout: VGPR r, lane l -> M=r+8*half, N=nl)
    const float* bias = z ? b_dst : b_src;
    const float bv = bias[n0 + nl];
    v8f acc;
#pragma unroll
    for (int r = 0; r < 8; ++r) acc[r] = bv;

    const unsigned short* arow = xb + (size_t)(m0 + nl) * DIM;
    const unsigned short* brow = wt + (size_t)z * DIM * DIM + (size_t)(n0 + nl) * DIM;

#pragma unroll
    for (int k0 = 0; k0 < DIM; k0 += 32) {
        // A 16x32 bf16: lane holds K = 8*half..+7 and 16+8*half..+7 of row m0+nl
        v8bf alo = *(const v8bf*)(arow + k0 + half * 8);
        v8bf ahi = *(const v8bf*)(arow + k0 + 16 + half * 8);
        // B 32x16 bf16 (column n0+nl of W == row of Wt): lane holds K = 8*half..+7 and 16+8*half..+7
        v8bf blo = *(const v8bf*)(brow + k0 + half * 8);
        v8bf bhi = *(const v8bf*)(brow + k0 + 16 + half * 8);
        v16bf A, B;
#pragma unroll
        for (int i = 0; i < 8; ++i) {
            A[i] = alo[i]; A[i + 8] = ahi[i];
            B[i] = blo[i]; B[i + 8] = bhi[i];
        }
        acc = __builtin_amdgcn_wmma_f32_16x16x32_bf16(
                  false, A, false, B, (short)0, acc, false, false);
    }

    // Epilogue: store proj tile + fused per-head score partials.
    float* proj = z ? proj_dst : proj_src;
    float* sbuf = z ? s_dst : s_src;
    const int h = wave >> 1;                       // head = n0/32
    const int d = ((wave & 1) << 4) + nl;          // dim within head
    const float av = attn[h * (2 * HD) + (z ? HD : 0) + d];

#pragma unroll
    for (int r = 0; r < 8; ++r) {
        const int m = m0 + r + (half << 3);
        proj[(size_t)m * DIM + n0 + nl] = acc[r];
        float p = acc[r] * av;
        // reduce over the 16 lanes holding this M-row (width-16 butterflies)
        p += __shfl_xor(p, 1, 16);
        p += __shfl_xor(p, 2, 16);
        p += __shfl_xor(p, 4, 16);
        p += __shfl_xor(p, 8, 16);
        if (nl == 0) atomicAdd(&sbuf[m * H + h], p);   // 2 waves share each head
    }
}

// ---------------------------------------------------------------------------
// Edge phase 1: leaky-relu + exp of per-edge scores, accumulate softmax
// denominators per (dst, head).
// ---------------------------------------------------------------------------
__global__ __launch_bounds__(256)
void k_edge_scores(const long long* __restrict__ ei,
                   const float* __restrict__ s_src, const float* __restrict__ s_dst,
                   float* __restrict__ exbuf, float* __restrict__ denom) {
    int e = blockIdx.x * 256 + threadIdx.x;
    if (e >= EE) return;
    const int src = (int)ei[e];
    const int dst = (int)ei[EE + e];
#pragma unroll
    for (int h = 0; h < H; ++h) {
        float sc = s_src[src * H + h] + s_dst[dst * H + h];
        sc = sc >= 0.0f ? sc : 0.2f * sc;
        float ex = __expf(sc);
        exbuf[(size_t)e * H + h] = ex;
        atomicAdd(&denom[dst * H + h], ex);
    }
}

// ---------------------------------------------------------------------------
// Edge phase 2: weighted gather/scatter. 64 threads per edge, 4 channels each;
// gathers hit L2 (proj_src is 51MB << 192MB L2), scatter via f32 atomics.
// ---------------------------------------------------------------------------
__global__ __launch_bounds__(256)
void k_aggregate(const long long* __restrict__ ei,
                 const float* __restrict__ proj_src,
                 const float* __restrict__ exbuf, const float* __restrict__ denom,
                 float* __restrict__ out) {
    long long t = (long long)blockIdx.x * 256 + threadIdx.x;
    if (t >= (long long)EE * 64) return;
    const int e  = (int)(t >> 6);
    const int q  = (int)(t & 63);
    const int c0 = q << 2;
    const int h  = q >> 3;
    const int src = (int)ei[e];
    const int dst = (int)ei[EE + e];
    const float alpha = exbuf[(size_t)e * H + h] / (denom[dst * H + h] + 1e-15f);
    const float4 p = *(const float4*)(proj_src + (size_t)src * DIM + c0);
    float* o = out + (size_t)dst * DIM + c0;
    atomicAdd(o + 0, alpha * p.x);
    atomicAdd(o + 1, alpha * p.y);
    atomicAdd(o + 2, alpha * p.z);
    atomicAdd(o + 3, alpha * p.w);
}

// ---------------------------------------------------------------------------
extern "C" void kernel_launch(void* const* d_in, const int* in_sizes, int n_in,
                              void* d_out, int out_size, void* d_ws, size_t ws_size,
                              hipStream_t stream) {
    const float*     x     = (const float*)d_in[0];
    const long long* ei    = (const long long*)d_in[1];   // int64 (2, E)
    const float*     W_src = (const float*)d_in[2];
    const float*     b_src = (const float*)d_in[3];
    const float*     W_dst = (const float*)d_in[4];
    const float*     b_dst = (const float*)d_in[5];
    const float*     attn  = (const float*)d_in[6];
    float*           out   = (float*)d_out;

    // Workspace layout
    char* ws = (char*)d_ws;
    size_t off = 0;
    float*          proj_src = (float*)(ws + off);          off += (size_t)NN * DIM * 4;
    float*          proj_dst = (float*)(ws + off);          off += (size_t)NN * DIM * 4;
    unsigned short* xb       = (unsigned short*)(ws + off); off += (size_t)NN * DIM * 2;
    unsigned short* wt       = (unsigned short*)(ws + off); off += (size_t)2 * DIM * DIM * 2;
    float*          s_src    = (float*)(ws + off);          off += (size_t)NN * H * 4;
    float*          s_dst    = (float*)(ws + off);          off += (size_t)NN * H * 4;
    float*          denom    = (float*)(ws + off);          off += (size_t)NN * H * 4;
    float*          exbuf    = (float*)(ws + off);          off += (size_t)EE * H * 4;

    // Prep (s_src, s_dst, denom are contiguous -> single zero pass)
    {
        int nz = 3 * NN * H;
        k_zero<<<(nz + 255) / 256, 256, 0, stream>>>(s_src, nz);
    }
    {
        int n4 = NN * DIM / 4;
        k_cvt_x<<<(n4 + 255) / 256, 256, 0, stream>>>(x, xb, n4);
        k_init_out<<<(n4 + 255) / 256, 256, 0, stream>>>(x, out, n4);
    }
    k_cvt_w<<<(2 * DIM * DIM + 255) / 256, 256, 0, stream>>>(W_src, W_dst, wt);

    // Fused WMMA GEMM + scores: 3125 row-tiles, z = {src, dst}
    {
        dim3 grid(NN / 16, 1, 2);
        k_gemm_proj<<<grid, 512, 0, stream>>>(xb, wt, b_src, b_dst, attn,
                                              proj_src, proj_dst, s_src, s_dst);
    }

    // Edge softmax numerators + denominators
    k_edge_scores<<<(EE + 255) / 256, 256, 0, stream>>>(ei, s_src, s_dst, exbuf, denom);

    // Weighted aggregation into out (= x + agg)
    {
        long long nt = (long long)EE * 64;
        int blocks = (int)((nt + 255) / 256);
        k_aggregate<<<blocks, 256, 0, stream>>>(ei, proj_src, exbuf, denom, out);
    }
}
